// SpatialChannelAttention_34153579937798
// MI455X (gfx1250) — matrix operand order repaired and checked by reading, not compiled, not run on hardware
//
#include <hip/hip_runtime.h>

#define B_  4
#define C_  256
#define H_  64
#define W_  64
#define HW_ 4096
#define D_  32      // c/8
#define CR_ 16      // c/16 (MLP hidden)

typedef __attribute__((ext_vector_type(16))) _Float16 v16h;
typedef __attribute__((ext_vector_type(2)))  __fp16   fp16x2;
typedef __attribute__((ext_vector_type(8)))  float    v8f;

union U16H { v16h v; uint4 u[2]; unsigned u32[8]; _Float16 h[16]; };
union PK   { fp16x2 h; unsigned u; };

// ---------------- Stage 1: per (b,c) mean & max over spatial ----------------
__global__ void k_chan_stats(const float* __restrict__ x, float* __restrict__ avg,
                             float* __restrict__ mx) {
  int bc = blockIdx.x;                 // b*C + c
  int t = threadIdx.x;                 // 256
  const float* p = x + (size_t)bc * HW_;
  float s = 0.f, m = -3.0e38f;
  for (int i = t; i < HW_; i += 256) { float v = p[i]; s += v; m = fmaxf(m, v); }
  __shared__ float ss[256], sm[256];
  ss[t] = s; sm[t] = m; __syncthreads();
  for (int o = 128; o > 0; o >>= 1) {
    if (t < o) { ss[t] += ss[t + o]; sm[t] = fmaxf(sm[t], sm[t + o]); }
    __syncthreads();
  }
  if (t == 0) { avg[bc] = ss[0] * (1.f / HW_); mx[bc] = sm[0]; }
}

// ---------------- Stage 2: channel MLP + sigmoid ----------------
__global__ void k_chan_mlp(const float* __restrict__ avg, const float* __restrict__ mx,
                           const float* __restrict__ w1, const float* __restrict__ w2,
                           float* __restrict__ att) {
  int b = blockIdx.x, t = threadIdx.x;  // 256 threads
  __shared__ float sa[C_], sm[C_], h[CR_];
  sa[t] = avg[b * C_ + t]; sm[t] = mx[b * C_ + t];
  __syncthreads();
  if (t < CR_) {
    float ha = 0.f, hm = 0.f;
    for (int c = 0; c < C_; ++c) { float w = w1[t * C_ + c]; ha += w * sa[c]; hm += w * sm[c]; }
    h[t] = fmaxf(ha, 0.f) + fmaxf(hm, 0.f);
  }
  __syncthreads();
  float o = 0.f;
  #pragma unroll
  for (int j = 0; j < CR_; ++j) o += w2[t * CR_ + j] * h[j];
  att[b * C_ + t] = 1.f / (1.f + __expf(-o));
}

// ---------------- Stage 3: x *= ch_att ; spatial mean/max over channels ----------------
__global__ void k_apply_ch(const float* __restrict__ x, const float* __restrict__ att,
                           float* __restrict__ x1, float* __restrict__ sAvg,
                           float* __restrict__ sMax) {
  int t = threadIdx.x;
  int b = blockIdx.x / (HW_ / 256);
  int p = (blockIdx.x % (HW_ / 256)) * 256 + t;
  __shared__ float a[C_];
  a[t] = att[b * C_ + t];
  __syncthreads();
  const float* xb = x  + (size_t)b * C_ * HW_ + p;
  float*       ob = x1 + (size_t)b * C_ * HW_ + p;
  float s = 0.f, m = -3.0e38f;
  for (int c = 0; c < C_; ++c) {
    float v = xb[(size_t)c * HW_] * a[c];
    ob[(size_t)c * HW_] = v;
    s += v; m = fmaxf(m, v);
  }
  sAvg[b * HW_ + p] = s * (1.f / C_);
  sMax[b * HW_ + p] = m;
}

// ---------------- Stage 4: 7x7 conv over [avg;max] + sigmoid ----------------
__global__ void k_spconv(const float* __restrict__ sAvg, const float* __restrict__ sMax,
                         const float* __restrict__ wsp, float* __restrict__ sfac) {
  int t = threadIdx.x;
  int b = blockIdx.x / (HW_ / 256);
  int p = (blockIdx.x % (HW_ / 256)) * 256 + t;
  int y = p / W_, xp = p % W_;
  float acc = 0.f;
  for (int ky = 0; ky < 7; ++ky) {
    int yy = y + ky - 3; if (yy < 0 || yy >= H_) continue;
    for (int kx = 0; kx < 7; ++kx) {
      int xx = xp + kx - 3; if (xx < 0 || xx >= W_) continue;
      int q = yy * W_ + xx;
      acc += wsp[ky * 7 + kx] * sAvg[b * HW_ + q] + wsp[49 + ky * 7 + kx] * sMax[b * HW_ + q];
    }
  }
  sfac[b * HW_ + p] = 1.f / (1.f + __expf(-acc));
}

// ---------------- Stage 5: x *= sigmoid(sp)  (in place) ----------------
__global__ void k_apply_sp(float* __restrict__ x1, const float* __restrict__ sfac) {
  unsigned idx = blockIdx.x * 256u + threadIdx.x;  // < 4*2^20
  unsigned p = idx & (HW_ - 1);
  unsigned b = idx >> 20;                          // C_*HW_ = 2^20
  x1[idx] *= sfac[b * HW_ + p];
}

// ---------------- Stage 6: 1x1 projections -> f16 outputs ----------------
// transposed==1: out[b][j][dout] (q_t,k_t);  transposed==0: out[b][o][HW] (v_h)
__global__ void k_proj(const float* __restrict__ x2, const float* __restrict__ wmat,
                       const float* __restrict__ bias, int dout,
                       _Float16* __restrict__ out, int transposed) {
  int b  = blockIdx.z;
  int ot = blockIdx.y;                       // c_out tile of 16
  int j  = blockIdx.x * 256 + threadIdx.x;   // pixel
  __shared__ float w[16 * C_];
  __shared__ float bsh[16];
  for (int i = threadIdx.x; i < 16 * C_; i += 256) w[i] = wmat[(size_t)(ot * 16) * C_ + i];
  if (threadIdx.x < 16) bsh[threadIdx.x] = bias[ot * 16 + threadIdx.x];
  __syncthreads();
  const float* xb = x2 + (size_t)b * C_ * HW_ + j;
  float acc[16];
  #pragma unroll
  for (int o = 0; o < 16; ++o) acc[o] = bsh[o];
  for (int c = 0; c < C_; ++c) {
    float xv = xb[(size_t)c * HW_];
    #pragma unroll
    for (int o = 0; o < 16; ++o) acc[o] = fmaf(w[o * C_ + c], xv, acc[o]);
  }
  if (transposed) {
    _Float16* ob = out + ((size_t)b * HW_ + j) * dout + ot * 16;
    #pragma unroll
    for (int o = 0; o < 16; ++o) ob[o] = (_Float16)acc[o];
  } else {
    _Float16* ob = out + ((size_t)b * dout + ot * 16) * HW_ + j;
    #pragma unroll
    for (int o = 0; o < 16; ++o) ob[(size_t)o * HW_] = (_Float16)acc[o];
  }
}

// ---------------- Stage 7: flash attention (transposed tiles) + residual ----------------
// S^T = K . Q^T  (A = K-tile 16j x 32d, B = Q-tile 32d x 16i) -> per-lane column i
// O^T = V . P^T  (A = V-tile 16c x 32j, B = P^T 32j x 16i)    -> accumulator N = i
// V tiles double-buffered in LDS via global_load_async_to_lds_b128 (ASYNCcnt).
__global__ __launch_bounds__(128)
void k_flash(const _Float16* __restrict__ qt, const _Float16* __restrict__ kt,
             const _Float16* __restrict__ vh, const float* __restrict__ x2,
             const float* __restrict__ gamma, float* __restrict__ out) {
  __shared__ _Float16 Vt[2][C_][40];    // two 32-j tiles, rows padded to 80B

  const int tid  = threadIdx.x;
  const int wave = tid >> 5;
  const int lane = tid & 31;
  const int hl   = lane >> 4;           // half-wave
  const int ln   = lane & 15;
  const int b    = blockIdx.y;
  const int i0   = blockIdx.x * 64 + wave * 16;

  const _Float16* kb_base = kt + (size_t)b * HW_ * D_;
  const _Float16* vb_base = vh + (size_t)b * C_ * HW_;

  // issue async V-tile load: 4 instructions per thread (512 x 16B chunks / 128 threads)
  auto issue_tile = [&](int jj, int nb) {
    #pragma unroll
    for (int k = 0; k < 4; ++k) {
      int idx = tid + k * 128;
      int c = idx >> 1, seg = idx & 1;
      unsigned laddr = (unsigned)(size_t)&Vt[nb][c][seg * 16];
      const _Float16* g = vb_base + (size_t)c * HW_ + jj + seg * 16;
      asm volatile("global_load_async_to_lds_b128 %0, %1, off"
                   :: "v"(laddr), "v"(g) : "memory");
    }
  };

  // Q B-operand: N = i = i0+ln, K = d per A-layout (two 16B chunks: hl*8, 16+hl*8)
  U16H qb;
  {
    const _Float16* qrow = qt + ((size_t)b * HW_ + (i0 + ln)) * D_;
    qb.u[0] = *(const uint4*)(qrow + hl * 8);
    qb.u[1] = *(const uint4*)(qrow + 16 + hl * 8);
  }

  v8f zero = {0.f, 0.f, 0.f, 0.f, 0.f, 0.f, 0.f, 0.f};
  v8f O[16];
  #pragma unroll
  for (int ct = 0; ct < 16; ++ct) O[ct] = zero;
  float mcur = -3.0e38f, lcur = 0.f;    // per-lane: state for column i = i0+ln

  issue_tile(0, 0);                     // prologue: tile 0 -> buffer 0

  for (int jj = 0; jj < HW_; jj += 32) {
    const int cur = (jj >> 5) & 1;
    if (jj + 32 < HW_) {
      issue_tile(jj + 32, cur ^ 1);     // prefetch next tile into other buffer
      asm volatile("s_wait_asynccnt 0x4" ::: "memory");  // current tile's 4 done
    } else {
      asm volatile("s_wait_asynccnt 0x0" ::: "memory");
    }
    __syncthreads();                    // all waves' portions of buf[cur] in LDS

    // K A-operands: M = j local = ln (+16), K = d per A-layout (two 16B chunks)
    U16H ka0, ka1;
    {
      const _Float16* kr0 = kb_base + (size_t)(jj + ln) * D_;
      const _Float16* kr1 = kb_base + (size_t)(jj + 16 + ln) * D_;
      ka0.u[0] = *(const uint4*)(kr0 + hl * 8);
      ka0.u[1] = *(const uint4*)(kr0 + 16 + hl * 8);
      ka1.u[0] = *(const uint4*)(kr1 + hl * 8);
      ka1.u[1] = *(const uint4*)(kr1 + 16 + hl * 8);
    }

    // S^T tiles: s_t[r] = S(i = i0+ln, j = jj + 16t + r + 8*hl)
    v8f s0 = __builtin_amdgcn_wmma_f32_16x16x32_f16(false, ka0.v, false, qb.v, (short)0, zero, false, false);
    v8f s1 = __builtin_amdgcn_wmma_f32_16x16x32_f16(false, ka1.v, false, qb.v, (short)0, zero, false, false);

    // online softmax over j: 16 in-lane values + xor-16 partner holds the rest
    float mloc = fmaxf(s0[0], s1[0]);
    #pragma unroll
    for (int r = 1; r < 8; ++r) mloc = fmaxf(mloc, fmaxf(s0[r], s1[r]));
    mloc = fmaxf(mloc, __shfl_xor(mloc, 16, 32));
    float mn = fmaxf(mcur, mloc);
    float sum = 0.f;
    #pragma unroll
    for (int r = 0; r < 8; ++r) {
      s0[r] = __expf(s0[r] - mn);
      s1[r] = __expf(s1[r] - mn);
      sum += s0[r] + s1[r];
    }
    sum += __shfl_xor(sum, 16, 32);
    float sc = __expf(mcur - mn);
    lcur = lcur * sc + sum;
    mcur = mn;
    #pragma unroll
    for (int ct = 0; ct < 16; ++ct) {
      #pragma unroll
      for (int r = 0; r < 8; ++r) O[ct][r] *= sc;   // sc uniform per lane
    }

    // Build P^T B-operand in registers: pack f16 pairs, one xor-16 exchange
    unsigned pk0[4], pk1[4];
    #pragma unroll
    for (int v = 0; v < 4; ++v) {
      PK a, c;
      a.h = __builtin_amdgcn_cvt_pkrtz(s0[2 * v], s0[2 * v + 1]); pk0[v] = a.u;
      c.h = __builtin_amdgcn_cvt_pkrtz(s1[2 * v], s1[2 * v + 1]); pk1[v] = c.u;
    }
    U16H pb;
    #pragma unroll
    for (int v = 0; v < 4; ++v) {
      unsigned send = hl ? pk0[v] : pk1[v];
      unsigned rc = __shfl_xor(send, 16, 32);
      pb.u32[v]     = hl ? rc     : pk0[v];
      pb.u32[v + 4] = hl ? pk1[v] : rc;
    }

    // O^T += V-tile (16c x 32j) . P^T (32j x 16i)
    #pragma unroll
    for (int ct = 0; ct < 16; ++ct) {
      U16H va;
      const _Float16* vr = &Vt[cur][ct * 16 + ln][0];
      va.u[0] = *(const uint4*)(vr + hl * 16);
      va.u[1] = *(const uint4*)(vr + hl * 16 + 8);
      O[ct] = __builtin_amdgcn_wmma_f32_16x16x32_f16(false, va.v, false, pb.v, (short)0, O[ct], false, false);
    }
    __syncthreads();                    // all waves done reading buf[cur]
  }

  // epilogue: out[b][c][i] = gamma * O^T(c,i)/l(i) + x2[b][c][i]  (i = fast axis)
  float g = gamma[0];
  float linv = 1.f / lcur;
  int i = i0 + ln;
  #pragma unroll
  for (int ct = 0; ct < 16; ++ct) {
    #pragma unroll
    for (int r = 0; r < 8; ++r) {
      int c = ct * 16 + r + 8 * hl;
      size_t idx = ((size_t)b * C_ + c) * HW_ + i;
      out[idx] = g * O[ct][r] * linv + x2[idx];
    }
  }
}

// ---------------- host launcher ----------------
extern "C" void kernel_launch(void* const* d_in, const int* in_sizes, int n_in,
                              void* d_out, int out_size, void* d_ws, size_t ws_size,
                              hipStream_t stream) {
  const float* x    = (const float*)d_in[0];
  const float* w1   = (const float*)d_in[1];
  const float* w2   = (const float*)d_in[2];
  const float* wsp  = (const float*)d_in[3];
  const float* wq   = (const float*)d_in[4];
  const float* bq   = (const float*)d_in[5];
  const float* wk   = (const float*)d_in[6];
  const float* bk   = (const float*)d_in[7];
  const float* wv   = (const float*)d_in[8];
  const float* bv   = (const float*)d_in[9];
  const float* gam  = (const float*)d_in[10];
  float* out = (float*)d_out;

  char* ws = (char*)d_ws;
  size_t off = 0;
  float* x2    = (float*)(ws + off); off += (size_t)B_ * C_ * HW_ * 4;   // 16 MB
  float* chAvg = (float*)(ws + off); off += B_ * C_ * 4;
  float* chMax = (float*)(ws + off); off += B_ * C_ * 4;
  float* chAtt = (float*)(ws + off); off += B_ * C_ * 4;
  float* spAvg = (float*)(ws + off); off += B_ * HW_ * 4;
  float* spMax = (float*)(ws + off); off += B_ * HW_ * 4;
  float* sfac  = (float*)(ws + off); off += B_ * HW_ * 4;
  _Float16* q_t = (_Float16*)(ws + off); off += (size_t)B_ * HW_ * D_ * 2;
  _Float16* k_t = (_Float16*)(ws + off); off += (size_t)B_ * HW_ * D_ * 2;
  _Float16* v_h = (_Float16*)(ws + off); off += (size_t)B_ * C_ * HW_ * 2;
  (void)ws_size; (void)in_sizes; (void)n_in; (void)out_size;

  k_chan_stats<<<B_ * C_, 256, 0, stream>>>(x, chAvg, chMax);
  k_chan_mlp<<<B_, 256, 0, stream>>>(chAvg, chMax, w1, w2, chAtt);
  k_apply_ch<<<B_ * (HW_ / 256), 256, 0, stream>>>(x, chAtt, x2, spAvg, spMax);
  k_spconv<<<B_ * (HW_ / 256), 256, 0, stream>>>(spAvg, spMax, wsp, sfac);
  k_apply_sp<<<(B_ * C_ * HW_) / 256, 256, 0, stream>>>(x2, sfac);

  k_proj<<<dim3(HW_ / 256, D_ / 16, B_),  256, 0, stream>>>(x2, wq, bq, D_,  q_t, 1);
  k_proj<<<dim3(HW_ / 256, D_ / 16, B_),  256, 0, stream>>>(x2, wk, bk, D_,  k_t, 1);
  k_proj<<<dim3(HW_ / 256, C_ / 16, B_),  256, 0, stream>>>(x2, wv, bv, C_,  v_h, 0);

  k_flash<<<dim3(HW_ / 64, B_), 128, 0, stream>>>(q_t, k_t, v_h, x2, gam, out);
}
